// PairwiseAttentionInitial_66984309948706
// MI455X (gfx1250) — matrix operand order repaired and checked by reading, not compiled, not run on hardware
//
#include <hip/hip_runtime.h>
#include <hip/hip_bf16.h>
#include <hip/hip_fp16.h>

typedef __attribute__((ext_vector_type(16))) _Float16 v16h;
typedef __attribute__((ext_vector_type(8)))  _Float16 v8h;
typedef __attribute__((ext_vector_type(8)))  float    v8f;
typedef __attribute__((__vector_size__(16))) int      vi4;   // for async-LDS builtin

#define LSEQ 256
#define FDIM 256
#define NH   8
#define HC   32
#define SQC  0.17677669529663687f   // 1/sqrt(32)

__device__ __forceinline__ v8f vzero8f() {
  v8f z;
#pragma unroll
  for (int i = 0; i < 8; ++i) z[i] = 0.f;
  return z;
}

// Build A fragment (16x32 f16, K = e + 16*hi) from a row pointer
// already offset by (+ 16*hi): two contiguous 16B chunks at +0 and +8 halves.
__device__ __forceinline__ v16h make_a(const _Float16* base) {
  v8h a0 = *(const v8h*)(base);
  v8h a1 = *(const v8h*)(base + 8);
  v16h a;
#pragma unroll
  for (int i = 0; i < 8; ++i) { a[i] = a0[i]; a[i + 8] = a1[i]; }
  return a;
}

__device__ __forceinline__ v8f wmma_f16(v16h a, v16h b, v8f c) {
  return __builtin_amdgcn_wmma_f32_16x16x32_f16(false, a, false, b,
                                                (short)0, c, false, false);
}

// 16-byte async global->LDS copy (CDNA5 GLOBAL_LOAD_ASYNC_TO_LDS_B128, tracked
// by ASYNCcnt). The builtin's pointer params are 'int __vector(4) *'.
// Guarded: falls back to a plain copy if the builtin is absent; the
// s_wait_asynccnt fences are harmless no-ops in that case.
__device__ __forceinline__ void cp_async16(const _Float16* g, _Float16* l) {
#if __has_builtin(__builtin_amdgcn_global_load_async_to_lds_b128)
  __builtin_amdgcn_global_load_async_to_lds_b128(
      (vi4*)const_cast<_Float16*>(g), (vi4*)l, 0, 0);
#else
  *(v8h*)l = *(const v8h*)g;
#endif
}

// ---------------------------------------------------------------------------
// Pack weight [256 x 256] f32 -> WMMA B-fragment order f16, kt-major so each
// K-slice (16 nt x 512 halves = 16KB) is contiguous for async LDS staging:
// idx = ((kt*16 + nt)*32 + lane)*16 + e ; element (k = kt*32 + e + 16*hi,
// n = nt*16 + lane&15). One 64K-element plane per weight.
__global__ void pack_w_kernel(const float* w0, const float* w1, const float* w2,
                              const float* w3, const float* w4, _Float16* dst) {
  int idx = blockIdx.x * 256 + threadIdx.x;
  const float* W;
  switch (blockIdx.y) {
    case 0: W = w0; break; case 1: W = w1; break; case 2: W = w2; break;
    case 3: W = w3; break; default: W = w4; break;
  }
  _Float16* D = dst + (size_t)blockIdx.y * 65536;
  int e = idx & 15, lane = (idx >> 4) & 31, nt = (idx >> 9) & 15, kt = idx >> 13;
  int n = nt * 16 + (lane & 15);
  int k = kt * 32 + e + 16 * (lane >> 4);
  D[idx] = (_Float16)W[k * 256 + n];
}

// ---------------------------------------------------------------------------
// LayerNorm over F: one wave per row, 8 floats/lane, shfl reduction. f16 out.
__global__ void ln_kernel(const float* __restrict__ x, const float* __restrict__ g,
                          const float* __restrict__ beta, _Float16* __restrict__ y) {
  int wave = threadIdx.x >> 5, lane = threadIdx.x & 31;
  int row = blockIdx.x * 8 + wave;
  const float4* xr4 = (const float4*)(x + (size_t)row * FDIM);
  float4 v0 = xr4[lane * 2], v1 = xr4[lane * 2 + 1];
  float xv[8] = {v0.x, v0.y, v0.z, v0.w, v1.x, v1.y, v1.z, v1.w};
  float s = 0.f, s2 = 0.f;
#pragma unroll
  for (int j = 0; j < 8; ++j) { s += xv[j]; s2 += xv[j] * xv[j]; }
#pragma unroll
  for (int m = 16; m >= 1; m >>= 1) {
    s  += __shfl_xor(s,  m, 32);
    s2 += __shfl_xor(s2, m, 32);
  }
  float mu  = s * (1.f / FDIM);
  float inv = rsqrtf(s2 * (1.f / FDIM) - mu * mu + 1e-5f);
  _Float16* yr = y + (size_t)row * FDIM;
#pragma unroll
  for (int j = 0; j < 8; ++j) {
    int f = lane * 8 + j;
    yr[f] = (_Float16)((xv[j] - mu) * inv * g[f] + beta[f]);
  }
}

// ---------------------------------------------------------------------------
// Projections: x_ln [65536 x 256] @ W[mode] (+bias). Grid x: (rowBlk<<1)|ntHalf.
// Block = 128 rows x 128 cols; wave owns 16 rows x 8 tiles (64 accum VGPRs).
// Weight K-slices (8KB) double-buffered in LDS via async copies.
// mode: 0=q, 1=k (row-major [B,S,H,C]); 2=v transposed [B,H,C,S]; 3=gate sigmoid.
__global__ __launch_bounds__(256, 2)
void proj_gemm_kernel(const _Float16* __restrict__ xln,
                      const _Float16* __restrict__ wpack,
                      const float* __restrict__ bq, const float* __restrict__ bk,
                      const float* __restrict__ bv, const float* __restrict__ bg,
                      _Float16* __restrict__ q, _Float16* __restrict__ kk,
                      _Float16* __restrict__ vT, _Float16* __restrict__ gate) {
  __shared__ __attribute__((aligned(32))) _Float16 bsh[2][4096]; // 2 x 8KB
  int mode = blockIdx.y;
  int nh = blockIdx.x & 1;
  int rowBlk = blockIdx.x >> 1;
  const _Float16* W = wpack + (size_t)mode * 65536 + nh * 4096;
  int tid = threadIdx.x;
  int w = tid >> 5, lane = tid & 31, hi = lane >> 4, lm = lane & 15;
  int rowBase = rowBlk * 128 + w * 16;
  const _Float16* arow = xln + (size_t)(rowBase + lm) * FDIM;
  v8f acc[8];
#pragma unroll
  for (int nt = 0; nt < 8; ++nt) acc[nt] = vzero8f();

  cp_async16(W + tid * 8,        bsh[0] + tid * 8);          // stage kt=0
  cp_async16(W + 2048 + tid * 8, bsh[0] + 2048 + tid * 8);
  for (int kt = 0; kt < 8; ++kt) {
    if (kt < 7) {                                            // prefetch kt+1
      const _Float16* Ws = W + (size_t)(kt + 1) * 8192;
      _Float16* dbuf = bsh[(kt + 1) & 1];
      cp_async16(Ws + tid * 8,        dbuf + tid * 8);
      cp_async16(Ws + 2048 + tid * 8, dbuf + 2048 + tid * 8);
      asm volatile("s_wait_asynccnt 0x2" ::: "memory");      // kt slice landed
    } else {
      asm volatile("s_wait_asynccnt 0x0" ::: "memory");
    }
    __syncthreads();
    v16h a = make_a(arow + kt * 32 + 16 * hi);
    const _Float16* lb = bsh[kt & 1];
#pragma unroll
    for (int nt = 0; nt < 8; ++nt) {
      v16h b = *(const v16h*)(lb + nt * 512 + lane * 16);
      acc[nt] = wmma_f16(a, b, acc[nt]);
    }
    __syncthreads();
  }
  const float* bias = (mode == 0) ? bq : (mode == 1) ? bk : (mode == 2) ? bv : bg;
#pragma unroll
  for (int nt = 0; nt < 8; ++nt) {
    int n = (nh * 8 + nt) * 16 + lm;
    float bval = bias[n];
#pragma unroll
    for (int r = 0; r < 8; ++r) {
      int R = rowBase + r + 8 * hi;     // D row = r + 8*hi
      float val = acc[nt][r] + bval;
      if (mode == 0)      q [(size_t)R * 256 + n] = (_Float16)val;
      else if (mode == 1) kk[(size_t)R * 256 + n] = (_Float16)val;
      else if (mode == 2) {
        int b_ = R >> 8, s_ = R & 255;  // R = b*L + s
        vT[(((size_t)b_ * NH + (n >> 5)) * HC + (n & 31)) * LSEQ + s_] = (_Float16)val;
      } else
        gate[(size_t)R * 256 + n] = (_Float16)(1.f / (1.f + __expf(-val)));
    }
  }
}

// ---------------------------------------------------------------------------
// bias = x_ln @ wb + bb -> stored [H][L*L] f32 so attention reads coalesce.
__global__ void bias_kernel(const _Float16* __restrict__ xln, const float* __restrict__ wb,
                            const float* __restrict__ bb, float* __restrict__ biasOut) {
  int idx = blockIdx.x * 256 + threadIdx.x;     // (r, h)
  int h = idx & 7, r = idx >> 3;
  const _Float16* xr = xln + (size_t)r * FDIM;
  float s = bb[h];
  for (int f = 0; f < FDIM; ++f) s += (float)xr[f] * wb[f * NH + h];
  biasOut[(size_t)h * 65536 + r] = s;
}

// ---------------------------------------------------------------------------
// Attention: one block per (b,h). K tile (16KB) and V tile (16KB) async-staged
// to LDS once; each wave runs two 16-row strips fully in registers:
// QK^T (16 WMMAs) -> softmax via shfl_xor -> f16 weights through a per-wave
// 8KB LDS strip (layout transpose) -> P@V (16 WMMAs) -> gate -> f16 store.
__global__ __launch_bounds__(256, 2)
void attn_kernel(const _Float16* __restrict__ q, const _Float16* __restrict__ k,
                 const _Float16* __restrict__ vT, const _Float16* __restrict__ gate,
                 const float* __restrict__ biasA, _Float16* __restrict__ attn) {
  extern __shared__ __attribute__((aligned(32))) _Float16 smem[];
  _Float16* ksh  = smem;           // 256 rows x 32 halves = 16KB
  _Float16* vsh  = smem + 8192;    // 32 rows x 256 halves = 16KB
  _Float16* wlds = smem + 16384;   // 8 waves x 4096 halves = 64KB
  int b = blockIdx.x >> 3, h = blockIdx.x & 7;
  int tid = threadIdx.x;
  int w = tid >> 5, lane = tid & 31, hi = lane >> 4, lm = lane & 15;
  const _Float16* qb = q  + (size_t)b * LSEQ * 256 + h * HC;
  const _Float16* kb = k  + (size_t)b * LSEQ * 256 + h * HC;
  const _Float16* vb = vT + ((size_t)b * NH + h) * HC * LSEQ;
  const float*    bh = biasA + (size_t)h * 65536;
  _Float16* wl = wlds + w * 4096;

  // ---- async stage K tile (row s -> ksh[s*32..]) and V tile (contiguous)
#pragma unroll
  for (int j = 0; j < 4; ++j)
    cp_async16(kb + (size_t)tid * 256 + j * 8, ksh + tid * 32 + j * 8);
#pragma unroll
  for (int j = 0; j < 4; ++j) {
    int c4 = tid + 256 * j;
    cp_async16(vb + (size_t)(c4 >> 5) * 256 + (c4 & 31) * 8, vsh + c4 * 8);
  }
  asm volatile("s_wait_asynccnt 0x0" ::: "memory");
  __syncthreads();

  for (int sIdx = 0; sIdx < 2; ++sIdx) {
    int mt = w + sIdx * 8;
    v8f acc[16];
    // ---- GEMM1: logits strip [16 x 256], single K=32 WMMA per tile
    v16h a = make_a(qb + (size_t)(mt * 16 + lm) * 256 + 16 * hi);
#pragma unroll
    for (int nt = 0; nt < 16; ++nt) {
      v16h bf = *(const v16h*)(ksh + (nt * 16 + lm) * 32 + 16 * hi);
      acc[nt] = wmma_f16(a, bf, vzero8f());
    }
    // ---- scale + pair bias + softmax over the 256 columns of each row
#pragma unroll
    for (int r = 0; r < 8; ++r) {
      int l = mt * 16 + r + 8 * hi;             // this lane's D row
      const float* br = bh + (size_t)l * 256;
      float mx = -3.0e38f;
#pragma unroll
      for (int nt = 0; nt < 16; ++nt) {
        float v_ = acc[nt][r] * SQC + br[nt * 16 + lm];
        acc[nt][r] = v_;
        mx = fmaxf(mx, v_);
      }
#pragma unroll
      for (int m = 8; m >= 1; m >>= 1) mx = fmaxf(mx, __shfl_xor(mx, m, 32));
      float sm = 0.f;
#pragma unroll
      for (int nt = 0; nt < 16; ++nt) {
        float e = __expf(acc[nt][r] - mx);
        acc[nt][r] = e;
        sm += e;
      }
#pragma unroll
      for (int m = 8; m >= 1; m >>= 1) sm += __shfl_xor(sm, m, 32);
      float inv = 1.f / sm;
#pragma unroll
      for (int nt = 0; nt < 16; ++nt)
        wl[(r + 8 * hi) * 256 + nt * 16 + lm] = (_Float16)(acc[nt][r] * inv);
    }
    asm volatile("s_wait_dscnt 0" ::: "memory");   // wave-local LDS transpose fence
    // ---- GEMM2: strip @ V -> [16 x 32], K = 256 (8 WMMAs per output tile)
#pragma unroll
    for (int nt = 0; nt < 2; ++nt) {
      v8f o = vzero8f();
#pragma unroll
      for (int kt = 0; kt < 8; ++kt) {
        v16h aw = make_a(wl + lm * 256 + kt * 32 + 16 * hi);
        v16h bf = *(const v16h*)(vsh + (nt * 16 + lm) * 256 + kt * 32 + 16 * hi);
        o = wmma_f16(aw, bf, o);
      }
#pragma unroll
      for (int r = 0; r < 8; ++r) {
        int l = mt * 16 + r + 8 * hi;
        int c = nt * 16 + lm;
        size_t gi = ((size_t)(b * LSEQ + l)) * 256 + h * HC + c;
        attn[gi] = (_Float16)(o[r] * (float)gate[gi]);
      }
    }
    asm volatile("s_wait_dscnt 0" ::: "memory");
  }
}

// ---------------------------------------------------------------------------
// Output projection: attn [65536 x 256] f16 @ wo (packed) + bo -> f32 out.
// Same LDS double-buffered weight staging as proj_gemm.
__global__ __launch_bounds__(256, 2)
void out_gemm_kernel(const _Float16* __restrict__ attn,
                     const _Float16* __restrict__ Wp,
                     const float* __restrict__ bo, float* __restrict__ out) {
  __shared__ __attribute__((aligned(32))) _Float16 bsh[2][4096];
  int nh = blockIdx.x & 1;
  int rowBlk = blockIdx.x >> 1;
  const _Float16* W = Wp + nh * 4096;
  int tid = threadIdx.x;
  int w = tid >> 5, lane = tid & 31, hi = lane >> 4, lm = lane & 15;
  int rowBase = rowBlk * 128 + w * 16;
  const _Float16* arow = attn + (size_t)(rowBase + lm) * 256;
  v8f acc[8];
#pragma unroll
  for (int nt = 0; nt < 8; ++nt) acc[nt] = vzero8f();

  cp_async16(W + tid * 8,        bsh[0] + tid * 8);
  cp_async16(W + 2048 + tid * 8, bsh[0] + 2048 + tid * 8);
  for (int kt = 0; kt < 8; ++kt) {
    if (kt < 7) {
      const _Float16* Ws = W + (size_t)(kt + 1) * 8192;
      _Float16* dbuf = bsh[(kt + 1) & 1];
      cp_async16(Ws + tid * 8,        dbuf + tid * 8);
      cp_async16(Ws + 2048 + tid * 8, dbuf + 2048 + tid * 8);
      asm volatile("s_wait_asynccnt 0x2" ::: "memory");
    } else {
      asm volatile("s_wait_asynccnt 0x0" ::: "memory");
    }
    __syncthreads();
    v16h a = make_a(arow + kt * 32 + 16 * hi);
    const _Float16* lb = bsh[kt & 1];
#pragma unroll
    for (int nt = 0; nt < 8; ++nt) {
      v16h b = *(const v16h*)(lb + nt * 512 + lane * 16);
      acc[nt] = wmma_f16(a, b, acc[nt]);
    }
    __syncthreads();
  }
#pragma unroll
  for (int nt = 0; nt < 8; ++nt) {
    int n = (nh * 8 + nt) * 16 + lm;
    float bb_ = bo[n];
#pragma unroll
    for (int r = 0; r < 8; ++r)
      out[(size_t)(rowBase + r + 8 * hi) * 256 + n] = acc[nt][r] + bb_;
  }
}

// ---------------------------------------------------------------------------
extern "C" void kernel_launch(void* const* d_in, const int* in_sizes, int n_in,
                              void* d_out, int out_size, void* d_ws, size_t ws_size,
                              hipStream_t stream) {
  (void)in_sizes; (void)n_in; (void)out_size; (void)ws_size;
  const float* features = (const float*)d_in[0];
  const float* ln_scale = (const float*)d_in[1];
  const float* ln_bias  = (const float*)d_in[2];
  const float* wq = (const float*)d_in[3];  const float* bq = (const float*)d_in[4];
  const float* wk = (const float*)d_in[5];  const float* bk = (const float*)d_in[6];
  const float* wv = (const float*)d_in[7];  const float* bv = (const float*)d_in[8];
  const float* wb = (const float*)d_in[9];  const float* bb = (const float*)d_in[10];
  const float* wg = (const float*)d_in[11]; const float* bg = (const float*)d_in[12];
  const float* wo = (const float*)d_in[13]; const float* bo = (const float*)d_in[14];
  float* out = (float*)d_out;

  char* base = (char*)d_ws;
  size_t off = 0;
  auto alloc = [&](size_t bytes) -> void* {
    void* p = base + off;
    off += (bytes + 255) & ~(size_t)255;
    return p;
  };
  const size_t NEL = (size_t)65536 * 256;        // 16.7M elements
  _Float16* xln   = (_Float16*)alloc(NEL * 2);
  _Float16* qbuf  = (_Float16*)alloc(NEL * 2);
  _Float16* kbuf  = (_Float16*)alloc(NEL * 2);
  _Float16* vTb   = (_Float16*)alloc(NEL * 2);
  _Float16* gateb = (_Float16*)alloc(NEL * 2);
  _Float16* attnb = (_Float16*)alloc(NEL * 2);
  float*    biasA = (float*)   alloc((size_t)65536 * 8 * 4);
  _Float16* wpack = (_Float16*)alloc((size_t)5 * 65536 * 2);

  pack_w_kernel<<<dim3(256, 5), 256, 0, stream>>>(wq, wk, wv, wg, wo, wpack);
  ln_kernel<<<8192, 256, 0, stream>>>(features, ln_scale, ln_bias, xln);
  proj_gemm_kernel<<<dim3(1024, 4), 256, 0, stream>>>(xln, wpack, bq, bk, bv, bg,
                                                      qbuf, kbuf, vTb, gateb);
  bias_kernel<<<2048, 256, 0, stream>>>(xln, wb, bb, biasA);
  attn_kernel<<<2048, 256, 98304, stream>>>(qbuf, kbuf, vTb, gateb, biasA, attnb);
  out_gemm_kernel<<<1024, 256, 0, stream>>>(attnb, wpack + (size_t)4 * 65536, bo, out);
}
